// GATModel_76055280877749
// MI455X (gfx1250) — matrix operand order repaired and checked by reading, not compiled, not run on hardware
//
#include <hip/hip_runtime.h>

typedef __attribute__((ext_vector_type(16))) _Float16 v16h;
typedef __attribute__((ext_vector_type(4)))  _Float16 v4h;
typedef __attribute__((ext_vector_type(8)))  float    v8f;

static constexpr int NN   = 50000;
static constexpr int EE   = 800000;
static constexpr int E2   = EE + NN;      // edges + self loops
static constexpr int DD   = 64;
static constexpr int EDIM = 16;
static constexpr int LL   = 3;
static constexpr int GG   = 256;
static constexpr int HH   = 32;
static constexpr float SLOPE = 0.2f;
static constexpr unsigned ORD_NEG_INF = 0x007FFFFFu;   // f2ord(-inf)

// ---- order-preserving float <-> uint for atomic max on signed floats ----
__device__ __forceinline__ unsigned f2ord(float f) {
  unsigned u = __float_as_uint(f);
  return (u & 0x80000000u) ? ~u : (u | 0x80000000u);
}
__device__ __forceinline__ float ord2f(unsigned u) {
  return (u & 0x80000000u) ? __uint_as_float(u ^ 0x80000000u) : __uint_as_float(~u);
}

__global__ void fill_f32(float* __restrict__ p, float v, int n) {
  int i = blockIdx.x * blockDim.x + threadIdx.x;
  if (i < n) p[i] = v;
}
__global__ void fill_u32(unsigned* __restrict__ p, unsigned v, int n) {
  int i = blockIdx.x * blockDim.x + threadIdx.x;
  if (i < n) p[i] = v;
}

// degree + segment-sum of incoming edge_attr (self-loop fill_value='mean')
__global__ void deg_loop_kernel(const int* __restrict__ dst, const float* __restrict__ eattr,
                                float* __restrict__ deg, float* __restrict__ loopsum) {
  int i = blockIdx.x * blockDim.x + threadIdx.x;
  if (i >= EE * 4) return;                 // 4 channels per thread
  int e = i >> 2, q = (i & 3) << 2;
  int d = dst[e];
  float4 v = *(const float4*)(eattr + (size_t)e * EDIM + q);
  float* ls = loopsum + (size_t)d * EDIM + q;
  atomicAdd(ls + 0, v.x);
  atomicAdd(ls + 1, v.y);
  atomicAdd(ls + 2, v.z);
  atomicAdd(ls + 3, v.w);
  if ((i & 3) == 0) atomicAdd(&deg[d], 1.0f);
}

// fold lin_edge into a 16-vector: ve[l][k] = sum_d We[l][k][d] * a_edge[l][d]
__global__ void ve_kernel(const float* __restrict__ We, const float* __restrict__ a_edge,
                          float* __restrict__ ve) {
  int t = blockIdx.x * blockDim.x + threadIdx.x;
  if (t >= LL * EDIM) return;
  int l = t / EDIM, k = t % EDIM;
  float s = 0.f;
  for (int d = 0; d < DD; ++d) s += We[(l * EDIM + k) * DD + d] * a_edge[l * DD + d];
  ve[t] = s;
}

// C[nrows x 64] = A[nrows x 64] * B[64 x 64]  via v_wmma_f32_16x16x32_f16
// 256 threads = 8 waves; block covers 32 rows; wave w -> C tile (w>>2)*16 rows, (w&3)*16 cols.
// A/B are staged in LDS in *fragment order*: each lane's v16h operand is 32 contiguous,
// 32B-aligned bytes -> two ds_load_b128 per operand, no v_mov_b16 repacking.
__global__ __launch_bounds__(256) void gemm64_wmma_kernel(
    const float* __restrict__ A, const float* __restrict__ B,
    float* __restrict__ C, int nrows) {
  __shared__ __align__(32) _Float16 sAf[2][2][32][16];  // [rowTile][kc][lane][j]  4 KB
  __shared__ __align__(32) _Float16 sBf[4][2][32][16];  // [colTile][kc][lane][j]  8 KB
  int t = threadIdx.x;
  int row0 = blockIdx.x * 32;

  // ---- A (32x64 fp32) -> f16 fragments; float4 global loads, v4h LDS stores ----
#pragma unroll
  for (int half = 0; half < 2; ++half) {
    int f = t + half * 256;    // float4 index, 512 total
    int r = f >> 4;            // 0..31
    int c4 = f & 15;
    int gr = row0 + r;
    float4 v = make_float4(0.f, 0.f, 0.f, 0.f);
    if (gr < nrows) v = ((const float4*)(A + (size_t)gr * DD))[c4];
    int c = c4 << 2;           // 0,4,..,60 (stays inside one 8-wide K group)
    int tile = r >> 4, lr = r & 15;
    int kc = c >> 5;
    int kk = c & 31;
    // 16-bit A 16x32 layout: lanes 0-15 hold K 0..7 & 16..23, lanes 16-31 hold K 8..15 & 24..31
    int lane, jb;
    if (kk < 8)       { lane = lr;      jb = kk; }
    else if (kk < 16) { lane = 16 + lr; jb = kk - 8; }
    else if (kk < 24) { lane = lr;      jb = 8 + (kk - 16); }
    else              { lane = 16 + lr; jb = 8 + (kk - 24); }
    v4h pk; pk[0] = (_Float16)v.x; pk[1] = (_Float16)v.y;
    pk[2] = (_Float16)v.z; pk[3] = (_Float16)v.w;
    *(v4h*)(&sAf[tile][kc][lane][jb]) = pk;
  }

  // ---- B (64x64 fp32) -> f16 fragments ----
  // thread t: column n = t&63, k block k4 = t>>6 (16 K values); coalesced row reads.
  {
    int n = t & 63;
    int k4 = t >> 6;           // 0..3
    int ntile = n >> 4, col = n & 15;
    int kc = k4 >> 1;
    int lane = ((k4 & 1) ? 16 : 0) + col;   // B 32x16: lanes 0-15 K 0..7 & 16..23, 16-31 K 8..15 & 24..31
    _Float16* dstp = &sBf[ntile][kc][lane][0];
#pragma unroll
    for (int q = 0; q < 4; ++q) {
      int kb = kc * 32 + (k4 & 1) * 8 + (q & 1) * 4 + (q >> 1) * 16;
      v4h pk;
      pk[0] = (_Float16)B[(kb + 0) * DD + n];
      pk[1] = (_Float16)B[(kb + 1) * DD + n];
      pk[2] = (_Float16)B[(kb + 2) * DD + n];
      pk[3] = (_Float16)B[(kb + 3) * DD + n];
      *(v4h*)(dstp + q * 4) = pk;
    }
  }
  __syncthreads();

  int wave = t >> 5, lane = t & 31;
  int mt = wave >> 2;          // row tile 0/1
  int nt = wave & 3;           // col tile 0..3
  v8f acc = {};
#pragma unroll
  for (int kc = 0; kc < 2; ++kc) {
    v16h a = *(const v16h*)(&sAf[mt][kc][lane][0]);
    v16h b = *(const v16h*)(&sBf[nt][kc][lane][0]);
    acc = __builtin_amdgcn_wmma_f32_16x16x32_f16(false, a, false, b, (short)0, acc,
                                                 false, false);
  }

  // C 16x16 f32 layout: lane -> col, vgpr r -> row r (+8 for lanes 16-31)
  int lr = lane & 15;
  int colg = nt * 16 + lr;
  int rbase = mt * 16 + ((lane >= 16) ? 8 : 0);
  float* Cp = C + (size_t)(row0 + rbase) * DD + colg;
  if (row0 + 32 <= nrows) {          // fast path: unguarded coalesced stores
#pragma unroll
    for (int r = 0; r < 8; ++r) Cp[r * DD] = acc[r];
  } else {
#pragma unroll
    for (int r = 0; r < 8; ++r)
      if (row0 + rbase + r < nrows) Cp[r * DD] = acc[r];
  }
}

// per-node attention scalars: ssrc = h . a_src, sdst = h . a_dst
__global__ void sdots_kernel(const float* __restrict__ h, const float* __restrict__ as,
                             const float* __restrict__ ad,
                             float* __restrict__ ssrc, float* __restrict__ sdst) {
  int i = blockIdx.x * blockDim.x + threadIdx.x;
  if (i >= NN) return;
  const float4* hv = (const float4*)(h + (size_t)i * DD);
  float s1 = 0.f, s2 = 0.f;
#pragma unroll
  for (int q = 0; q < 16; ++q) {
    float4 v = hv[q];
    s1 += v.x * as[q * 4 + 0] + v.y * as[q * 4 + 1] + v.z * as[q * 4 + 2] + v.w * as[q * 4 + 3];
    s2 += v.x * ad[q * 4 + 0] + v.y * ad[q * 4 + 1] + v.z * ad[q * 4 + 2] + v.w * ad[q * 4 + 3];
  }
  ssrc[i] = s1;
  sdst[i] = s2;
}

// edge logits + leaky relu + segment max (ordered-uint atomic max)
__global__ void attn_logit_kernel(const int* __restrict__ src, const int* __restrict__ dst,
                                  const float* __restrict__ eattr,
                                  const float* __restrict__ loopsum, const float* __restrict__ deg,
                                  const float* __restrict__ ve,
                                  const float* __restrict__ ssrc, const float* __restrict__ sdst,
                                  unsigned* __restrict__ m_ord, float* __restrict__ alog) {
  int i = blockIdx.x * blockDim.x + threadIdx.x;
  if (i >= E2) return;
  int s, d;
  float ed = 0.f;
  if (i < EE) {
    s = src[i]; d = dst[i];
    const float4* ea = (const float4*)(eattr + (size_t)i * EDIM);
#pragma unroll
    for (int q = 0; q < 4; ++q) {
      float4 v = ea[q];
      ed += v.x * ve[q * 4 + 0] + v.y * ve[q * 4 + 1] + v.z * ve[q * 4 + 2] + v.w * ve[q * 4 + 3];
    }
  } else {
    int nidx = i - EE; s = nidx; d = nidx;   // self loop, attr = mean of incoming
    const float4* ls = (const float4*)(loopsum + (size_t)nidx * EDIM);
#pragma unroll
    for (int q = 0; q < 4; ++q) {
      float4 v = ls[q];
      ed += v.x * ve[q * 4 + 0] + v.y * ve[q * 4 + 1] + v.z * ve[q * 4 + 2] + v.w * ve[q * 4 + 3];
    }
    ed /= fmaxf(deg[nidx], 1.0f);
  }
  float al = ssrc[s] + sdst[d] + ed;
  al = (al > 0.0f) ? al : SLOPE * al;
  alog[i] = al;
  atomicMax(&m_ord[d], f2ord(al));
}

__global__ void exp_denom_kernel(const int* __restrict__ dst, const unsigned* __restrict__ m_ord,
                                 float* __restrict__ alog, float* __restrict__ denom) {
  int i = blockIdx.x * blockDim.x + threadIdx.x;
  if (i >= E2) return;
  int d = (i < EE) ? dst[i] : (i - EE);
  float ex = expf(alog[i] - ord2f(m_ord[d]));
  alog[i] = ex;
  atomicAdd(&denom[d], ex);
}

__global__ void coeff_kernel(const int* __restrict__ dst, const float* __restrict__ denom,
                             float* __restrict__ alog) {
  int i = blockIdx.x * blockDim.x + threadIdx.x;
  if (i >= E2) return;
  int d = (i < EE) ? dst[i] : (i - EE);
  alog[i] = alog[i] / (denom[d] + 1e-16f);
}

// out[dst] += coeff[e] * h[src]  — float4 gather (b128), 16 threads/edge
__global__ void aggregate_kernel(const int* __restrict__ src, const int* __restrict__ dst,
                                 const float* __restrict__ coeff, const float* __restrict__ h,
                                 float* __restrict__ outb) {
  int i = blockIdx.x * blockDim.x + threadIdx.x;
  if (i >= E2 * 16) return;
  int e = i >> 4, q = (i & 15) << 2;
  int s = (e < EE) ? src[e] : (e - EE);
  int d = (e < EE) ? dst[e] : (e - EE);
  float cf = coeff[e];
  float4 hv = *(const float4*)(h + (size_t)s * DD + q);
  float* ob = outb + (size_t)d * DD + q;
  atomicAdd(ob + 0, hv.x * cf);
  atomicAdd(ob + 1, hv.y * cf);
  atomicAdd(ob + 2, hv.z * cf);
  atomicAdd(ob + 3, hv.w * cf);
}

__global__ void bias_relu_kernel(float* __restrict__ outb, const float* __restrict__ bias) {
  int i = blockIdx.x * blockDim.x + threadIdx.x;
  if (i >= NN * 16) return;
  int q = (i & 15) << 2;
  float4* p = (float4*)(outb + ((size_t)(i >> 4) * DD + q));
  float4 v = *p;
  v.x = fmaxf(v.x + bias[q + 0], 0.f);
  v.y = fmaxf(v.y + bias[q + 1], 0.f);
  v.z = fmaxf(v.z + bias[q + 2], 0.f);
  v.w = fmaxf(v.w + bias[q + 3], 0.f);
  *p = v;
}

__global__ void pool_kernel(const float* __restrict__ xf, const int* __restrict__ batch,
                            unsigned* __restrict__ gmax, float* __restrict__ gsum,
                            float* __restrict__ gcnt) {
  int i = blockIdx.x * blockDim.x + threadIdx.x;
  if (i >= NN * 16) return;
  int nidx = i >> 4, q = (i & 15) << 2;
  int g = batch[nidx];
  float4 v = *(const float4*)(xf + (size_t)nidx * DD + q);
  unsigned* gm = gmax + (size_t)g * DD + q;
  float*    gs = gsum + (size_t)g * DD + q;
  atomicMax(gm + 0, f2ord(v.x)); atomicAdd(gs + 0, v.x);
  atomicMax(gm + 1, f2ord(v.y)); atomicAdd(gs + 1, v.y);
  atomicMax(gm + 2, f2ord(v.z)); atomicAdd(gs + 2, v.z);
  atomicMax(gm + 3, f2ord(v.w)); atomicAdd(gs + 3, v.w);
  if ((i & 15) == 0) atomicAdd(&gcnt[g], 1.0f);
}

// [G,128] -> relu([G,32]) -> [G,1]; one 32-thread wave per graph
__global__ void mlp_kernel(const unsigned* __restrict__ gmax, const float* __restrict__ gsum,
                           const float* __restrict__ gcnt,
                           const float* __restrict__ W1, const float* __restrict__ b1,
                           const float* __restrict__ W2, const float* __restrict__ b2,
                           float* __restrict__ out) {
  int g = blockIdx.x;
  int j = threadIdx.x;            // 0..31
  __shared__ float hb[HH];
  float cnt = gcnt[g];
  float acc = b1[j];
  for (int k = 0; k < DD; ++k)
    acc += ord2f(gmax[g * DD + k]) * W1[k * HH + j];
  for (int k = 0; k < DD; ++k)
    acc += (gsum[g * DD + k] / cnt) * W1[(DD + k) * HH + j];
  hb[j] = fmaxf(acc, 0.0f) * W2[j];
  __syncthreads();
  if (j == 0) {
    float s = b2[0];
    for (int k = 0; k < HH; ++k) s += hb[k];
    out[g] = s;
  }
}

static inline int nblk(long long n, int bs) { return (int)((n + bs - 1) / bs); }

extern "C" void kernel_launch(void* const* d_in, const int* in_sizes, int n_in,
                              void* d_out, int out_size, void* d_ws, size_t ws_size,
                              hipStream_t stream) {
  const float* x      = (const float*)d_in[0];
  const int*   ei     = (const int*)d_in[1];
  const float* eattr  = (const float*)d_in[2];
  const int*   batch  = (const int*)d_in[3];
  const float* W      = (const float*)d_in[4];
  const float* We     = (const float*)d_in[5];
  const float* a_src  = (const float*)d_in[6];
  const float* a_dst  = (const float*)d_in[7];
  const float* a_edge = (const float*)d_in[8];
  const float* bias   = (const float*)d_in[9];
  const float* W1     = (const float*)d_in[10];
  const float* b1     = (const float*)d_in[11];
  const float* W2     = (const float*)d_in[12];
  const float* b2     = (const float*)d_in[13];
  float* out = (float*)d_out;

  const int* src = ei;
  const int* dst = ei + EE;

  // ---- carve workspace ----
  char* ws = (char*)d_ws;
  size_t off = 0;
  auto carve = [&](size_t bytes) -> char* {
    char* p = ws + off;
    off += (bytes + 255) & ~(size_t)255;
    return p;
  };
  float*    h       = (float*)carve((size_t)NN * DD * 4);
  float*    bufA    = (float*)carve((size_t)NN * DD * 4);
  float*    bufB    = (float*)carve((size_t)NN * DD * 4);
  float*    ssrc    = (float*)carve((size_t)NN * 4);
  float*    sdst    = (float*)carve((size_t)NN * 4);
  float*    deg     = (float*)carve((size_t)NN * 4);
  float*    loopsum = (float*)carve((size_t)NN * EDIM * 4);
  unsigned* m_ord   = (unsigned*)carve((size_t)NN * 4);
  float*    denom   = (float*)carve((size_t)NN * 4);
  float*    alog    = (float*)carve((size_t)E2 * 4);
  float*    ve      = (float*)carve((size_t)LL * EDIM * 4);
  unsigned* gmax    = (unsigned*)carve((size_t)GG * DD * 4);
  float*    gsum    = (float*)carve((size_t)GG * DD * 4);
  float*    gcnt    = (float*)carve((size_t)GG * 4);

  const int BS = 256;

  // self-loop attrs: degree + incoming edge_attr sums
  fill_f32<<<nblk(NN, BS), BS, 0, stream>>>(deg, 0.0f, NN);
  fill_f32<<<nblk((long long)NN * EDIM, BS), BS, 0, stream>>>(loopsum, 0.0f, NN * EDIM);
  deg_loop_kernel<<<nblk((long long)EE * 4, BS), BS, 0, stream>>>(dst, eattr, deg, loopsum);

  // folded edge projection vectors for all layers
  ve_kernel<<<1, 64, 0, stream>>>(We, a_edge, ve);

  const float* xcur = x;
  float* outb = bufA;
  for (int l = 0; l < LL; ++l) {
    outb = (l == 1) ? bufB : bufA;

    // h = xcur @ W[l]   (WMMA)
    gemm64_wmma_kernel<<<nblk(NN, 32), 256, 0, stream>>>(xcur, W + (size_t)l * DD * DD, h, NN);
    sdots_kernel<<<nblk(NN, BS), BS, 0, stream>>>(h, a_src + l * DD, a_dst + l * DD, ssrc, sdst);

    fill_u32<<<nblk(NN, BS), BS, 0, stream>>>(m_ord, ORD_NEG_INF, NN);
    fill_f32<<<nblk(NN, BS), BS, 0, stream>>>(denom, 0.0f, NN);
    fill_f32<<<nblk((long long)NN * DD, BS), BS, 0, stream>>>(outb, 0.0f, NN * DD);

    attn_logit_kernel<<<nblk(E2, BS), BS, 0, stream>>>(src, dst, eattr, loopsum, deg,
                                                       ve + l * EDIM, ssrc, sdst, m_ord, alog);
    exp_denom_kernel<<<nblk(E2, BS), BS, 0, stream>>>(dst, m_ord, alog, denom);
    coeff_kernel<<<nblk(E2, BS), BS, 0, stream>>>(dst, denom, alog);
    aggregate_kernel<<<nblk((long long)E2 * 16, BS), BS, 0, stream>>>(src, dst, alog, h, outb);
    bias_relu_kernel<<<nblk((long long)NN * 16, BS), BS, 0, stream>>>(outb, bias + l * DD);

    xcur = outb;
  }

  // global max+mean pooling, then MLP
  fill_u32<<<nblk((long long)GG * DD, BS), BS, 0, stream>>>(gmax, ORD_NEG_INF, GG * DD);
  fill_f32<<<nblk((long long)GG * DD, BS), BS, 0, stream>>>(gsum, 0.0f, GG * DD);
  fill_f32<<<nblk(GG, BS), BS, 0, stream>>>(gcnt, 0.0f, GG);
  pool_kernel<<<nblk((long long)NN * 16, BS), BS, 0, stream>>>(xcur, batch, gmax, gsum, gcnt);
  mlp_kernel<<<GG, HH, 0, stream>>>(gmax, gsum, gcnt, W1, b1, W2, b2, out);
}
